// TransformerBlock_850403525159
// MI455X (gfx1250) — matrix operand order repaired and checked by reading, not compiled, not run on hardware
//
#include <hip/hip_runtime.h>

// ---------------- problem constants ----------------
#define BATCH 512
#define SEQ   128
#define CDIM  192
#define NHEAD 6
#define HSIZE 32
#define CFF   768
#define NTOK  (BATCH * SEQ)          // 65536
#define EPS   1e-5f

typedef __bf16 bf16_t;
typedef __attribute__((ext_vector_type(16))) __bf16 v16bf;
typedef __attribute__((ext_vector_type(8)))  float  v8f;

// ---------------- WMMA fragment loaders (contiguous k order, same in A and B) ----
// A: 16(M) x 32(K) bf16. lane&15 = M, lane>>4 selects K-half offset (0/16).
// slot i: K = koff + i  -> 16 contiguous bf16 per lane.
__device__ __forceinline__ v16bf load_a_frag(const bf16_t* __restrict__ A,
                                             int lda, int kbase) {
  int lane = threadIdx.x & 31;
  int m    = lane & 15;
  int koff = (lane >> 4) << 4;              // 0 or 16
  const bf16_t* p = A + (size_t)m * lda + kbase + koff;
  v16bf a;
#pragma unroll
  for (int i = 0; i < 16; ++i) a[i] = p[i];
  return a;
}

// B: 32(K) x 16(N) bf16, sourced from an N-major buffer BT[n][k] (ldb = K
// stride of BT). lane&15 = N, lane>>4 selects K-half (0/16).
// slot i: K = koff + i  -> 16 contiguous bf16 (32B) per lane.
__device__ __forceinline__ v16bf load_bt_frag(const bf16_t* __restrict__ BT,
                                              int ldb, int kbase) {
  int lane = threadIdx.x & 31;
  int n    = lane & 15;
  int koff = (lane >> 4) << 4;              // 0 or 16
  const bf16_t* p = BT + (size_t)n * ldb + kbase + koff;
  v16bf b;
#pragma unroll
  for (int i = 0; i < 16; ++i) b[i] = p[i];
  return b;
}

__device__ __forceinline__ v8f wmma_bf16(v16bf a, v16bf b, v8f c) {
  return __builtin_amdgcn_wmma_f32_16x16x32_bf16(false, a, false, b,
                                                 (short)0, c, false, false);
}

// ---------------- weight prep: f32 -> bf16, N-major transpose -------------
// Wq/Wk/Wv: [H, C, HS] -> WT[n = h*HS+d][c] (n-major, ld = C)
__global__ void k_prep_qkvw(const float* __restrict__ Wq,
                            const float* __restrict__ Wk,
                            const float* __restrict__ Wv,
                            bf16_t* __restrict__ wqT,
                            bf16_t* __restrict__ wkT,
                            bf16_t* __restrict__ wvT) {
  int idx = blockIdx.x * blockDim.x + threadIdx.x;
  if (idx >= 3 * CDIM * CDIM) return;
  int which = idx / (CDIM * CDIM);
  int rem   = idx % (CDIM * CDIM);
  int n = rem / CDIM, c = rem % CDIM;
  int h = n / HSIZE, d = n % HSIZE;
  const float* W = (which == 0) ? Wq : (which == 1) ? Wk : Wv;
  bf16_t*      O = (which == 0) ? wqT : (which == 1) ? wkT : wvT;
  O[(size_t)n * CDIM + c] = (bf16_t)W[((size_t)h * CDIM + c) * HSIZE + d];
}

// Generic: W[K,N] f32 -> WT[n][k] bf16
__global__ void k_prep_wT(const float* __restrict__ W, bf16_t* __restrict__ WT,
                          int K, int N) {
  int idx = blockIdx.x * blockDim.x + threadIdx.x;
  if (idx >= K * N) return;
  int n = idx / K, k = idx % K;
  WT[(size_t)n * K + k] = (bf16_t)W[(size_t)k * N + n];
}

// ---------------- LayerNorm (f32 in -> bf16 out), wave per row ------------
__global__ void __launch_bounds__(256)
k_layernorm(const float* __restrict__ x, const float* __restrict__ g,
            const float* __restrict__ be, bf16_t* __restrict__ out) {
  int row  = blockIdx.x * 8 + (threadIdx.x >> 5);
  int lane = threadIdx.x & 31;
  const float* xr = x + (size_t)row * CDIM;
  float v[6];
  float s = 0.f;
#pragma unroll
  for (int j = 0; j < 6; ++j) { v[j] = xr[lane + 32 * j]; s += v[j]; }
#pragma unroll
  for (int m = 16; m >= 1; m >>= 1) s += __shfl_xor(s, m, 32);
  float mu = s * (1.0f / CDIM);
  float s2 = 0.f;
#pragma unroll
  for (int j = 0; j < 6; ++j) { float d = v[j] - mu; s2 += d * d; }
#pragma unroll
  for (int m = 16; m >= 1; m >>= 1) s2 += __shfl_xor(s2, m, 32);
  float rs = rsqrtf(s2 * (1.0f / CDIM) + EPS);
  bf16_t* orow = out + (size_t)row * CDIM;
#pragma unroll
  for (int j = 0; j < 6; ++j) {
    int idx = lane + 32 * j;
    orow[idx] = (bf16_t)((v[j] - mu) * rs * g[idx] + be[idx]);
  }
}

// ---------------- tiled WMMA GEMM: out[M,N] = A[M,K] * BT[N,K]^T ----------
// block: 256 thr = 8 waves -> 128(M) x 64(N) tile; wave = 16(M) x 64(N).
template <bool OUT_BF16, bool GELU, bool RES>
__global__ void __launch_bounds__(256)
k_gemm(const bf16_t* __restrict__ A, const bf16_t* __restrict__ BT,
       const float* __restrict__ bias, const float* __restrict__ res,
       void* __restrict__ outp, int N, int K) {
  int wave   = threadIdx.x >> 5;
  size_t m0  = (size_t)blockIdx.x * 128 + wave * 16;
  int n0     = blockIdx.y * 64;
  const bf16_t* Ap = A + m0 * K;
  v8f acc[4] = {};
  for (int kb = 0; kb < K; kb += 32) {
    v16bf a = load_a_frag(Ap, K, kb);
#pragma unroll
    for (int nt = 0; nt < 4; ++nt) {
      v16bf b = load_bt_frag(BT + (size_t)(n0 + nt * 16) * K, K, kb);
      acc[nt] = wmma_bf16(a, b, acc[nt]);
    }
  }
  // C/D layout: VGPR r -> M = r + 8*(lane>=16), N = lane&15
  int lane = threadIdx.x & 31;
  int n    = lane & 15;
  int mb   = (lane >> 4) * 8;
#pragma unroll
  for (int nt = 0; nt < 4; ++nt) {
    int col  = n0 + nt * 16 + n;
    float bv = bias ? bias[col] : 0.f;
#pragma unroll
    for (int r = 0; r < 8; ++r) {
      size_t m  = m0 + mb + r;
      float val = acc[nt][r] + bv;
      if (GELU) val = 0.5f * val * (1.0f + erff(val * 0.70710678118654752f));
      if (RES)  val += res[m * N + col];
      if (OUT_BF16) ((bf16_t*)outp)[m * N + col] = (bf16_t)val;
      else          ((float*)outp)[m * N + col]  = val;
    }
  }
}

// ---------------- fused attention: one workgroup per (b,h) ----------------
// S = Q K^T (WMMA) -> causal softmax (f32 LDS) -> P bf16 -> O = P V (WMMA)
#define SPITCH 132            // f32 pitch: 528B stride, bank-conflict free
#define VPITCH 136            // bf16 pitch: 272B stride, 16B aligned rows
#define ATT_SMEM (SEQ * SPITCH * 4 + SEQ * SEQ * 2 + HSIZE * VPITCH * 2)

__global__ void __launch_bounds__(256)
k_attention(const bf16_t* __restrict__ q, const bf16_t* __restrict__ k,
            const bf16_t* __restrict__ v, bf16_t* __restrict__ o) {
  extern __shared__ __align__(16) char smem[];
  float*  S  = (float*)smem;                                   // [128][SPITCH]
  bf16_t* P  = (bf16_t*)(smem + SEQ * SPITCH * 4);             // [128][128]
  bf16_t* Vt = (bf16_t*)(smem + SEQ * SPITCH * 4 + SEQ * SEQ * 2); // [32][VPITCH]

  int bh = blockIdx.x;
  int b  = bh / NHEAD, h = bh % NHEAD;
  const size_t base = (size_t)b * SEQ * CDIM + h * HSIZE;
  int tid  = threadIdx.x;
  int wave = tid >> 5;
  int lane = tid & 31;

  // stage V^T into LDS (coalesced: consecutive tid -> consecutive d)
  for (int i = tid; i < SEQ * HSIZE; i += 256) {
    int s = i >> 5, d = i & 31;
    Vt[d * VPITCH + s] = v[base + (size_t)s * CDIM + d];
  }

  // S = Q K^T : each wave owns one 16-row strip; K-dim = HS = 32 (one WMMA)
  {
    int m0 = wave * 16;
    v16bf a = load_a_frag(q + base + (size_t)m0 * CDIM, CDIM, 0);
    int n = lane & 15, mb = (lane >> 4) * 8;
#pragma unroll
    for (int nt = 0; nt < 8; ++nt) {
      v16bf bf = load_bt_frag(k + base + (size_t)(nt * 16) * CDIM, CDIM, 0);
      v8f acc = {};
      acc = wmma_bf16(a, bf, acc);
#pragma unroll
      for (int r = 0; r < 8; ++r)
        S[(m0 + mb + r) * SPITCH + nt * 16 + n] = acc[r];
    }
  }
  __syncthreads();

  // causal softmax, one thread per query row
  if (tid < SEQ) {
    int t = tid;
    float* Sr = S + t * SPITCH;
    const float scale = 0.17677669529663687f;   // 1/sqrt(32)
    float mx = -3.0e38f;
    for (int s2 = 0; s2 <= t; ++s2) {
      float val = Sr[s2] * scale;
      Sr[s2] = val;
      mx = fmaxf(mx, val);
    }
    float sum = 0.f;
    for (int s2 = 0; s2 <= t; ++s2) {
      float e = __expf(Sr[s2] - mx);
      Sr[s2] = e;
      sum += e;
    }
    float inv = 1.0f / sum;
    bf16_t* Pr = P + t * SEQ;
    for (int s2 = 0; s2 <= t; ++s2) Pr[s2] = (bf16_t)(Sr[s2] * inv);
    for (int s2 = t + 1; s2 < SEQ; ++s2) Pr[s2] = (bf16_t)0.f;
  }
  __syncthreads();

  // O = P V : M=128 (wave strip), N=32 (2 tiles), K=128 (4 steps)
  {
    int m0 = wave * 16;
    v8f acc0 = {}, acc1 = {};
    for (int kb = 0; kb < SEQ; kb += 32) {
      v16bf a  = load_a_frag(P + (size_t)m0 * SEQ, SEQ, kb);
      v16bf b0 = load_bt_frag(Vt, VPITCH, kb);
      v16bf b1 = load_bt_frag(Vt + 16 * VPITCH, VPITCH, kb);
      acc0 = wmma_bf16(a, b0, acc0);
      acc1 = wmma_bf16(a, b1, acc1);
    }
    int n = lane & 15, mb = (lane >> 4) * 8;
#pragma unroll
    for (int r = 0; r < 8; ++r) {
      size_t m = m0 + mb + r;
      o[base + m * CDIM + n]      = (bf16_t)acc0[r];
      o[base + m * CDIM + 16 + n] = (bf16_t)acc1[r];
    }
  }
}

// ---------------- host launcher -------------------------------------------
extern "C" void kernel_launch(void* const* d_in, const int* in_sizes, int n_in,
                              void* d_out, int out_size, void* d_ws, size_t ws_size,
                              hipStream_t stream) {
  const float* x   = (const float*)d_in[0];
  const float* Wq  = (const float*)d_in[1];
  const float* Wk  = (const float*)d_in[2];
  const float* Wv  = (const float*)d_in[3];
  const float* Wo  = (const float*)d_in[4];
  const float* bo  = (const float*)d_in[5];
  const float* W1  = (const float*)d_in[6];
  const float* b1  = (const float*)d_in[7];
  const float* W2  = (const float*)d_in[8];
  const float* b2  = (const float*)d_in[9];
  const float* g1  = (const float*)d_in[10];
  const float* be1 = (const float*)d_in[11];
  const float* g2  = (const float*)d_in[12];
  const float* be2 = (const float*)d_in[13];

  char* ws = (char*)d_ws;
  // bf16 weight region (N-major)
  const size_t WSZ  = (size_t)CDIM * CDIM * 2;        // 73728
  const size_t WFFZ = (size_t)CDIM * CFF * 2;         // 294912
  bf16_t* wqT = (bf16_t*)(ws + 0 * WSZ);
  bf16_t* wkT = (bf16_t*)(ws + 1 * WSZ);
  bf16_t* wvT = (bf16_t*)(ws + 2 * WSZ);
  bf16_t* woT = (bf16_t*)(ws + 3 * WSZ);
  bf16_t* w1T = (bf16_t*)(ws + 4 * WSZ);
  bf16_t* w2T = (bf16_t*)(ws + 4 * WSZ + WFFZ);
  size_t off = (size_t)1 << 20;                       // 1MB weight region

  const size_t ACT_BF = (size_t)NTOK * CDIM * 2;      // 25.2 MB
  const size_t ACT_F  = (size_t)NTOK * CDIM * 4;      // 50.3 MB
  const size_t FF_BF  = (size_t)NTOK * CFF * 2;       // 100.7 MB
  bf16_t* h1 = (bf16_t*)(ws + off);                   // also reused for o
  bf16_t* ob = h1;
  bf16_t* qb = (bf16_t*)(ws + off + ACT_BF);          // reused for h2
  bf16_t* h2 = qb;
  bf16_t* kb = (bf16_t*)(ws + off + 2 * ACT_BF);      // a overlays k,v,...
  bf16_t* vb = (bf16_t*)(ws + off + 3 * ACT_BF);
  bf16_t* ab = kb;                                    // [NTOK, CFF] bf16
  float*  x2 = (float*)(ws + off + 2 * ACT_BF + FF_BF);
  (void)ws_size; (void)n_in; (void)in_sizes; (void)out_size;

  // 1) weight conversion / transpose
  k_prep_qkvw<<<(3 * CDIM * CDIM + 255) / 256, 256, 0, stream>>>(
      Wq, Wk, Wv, wqT, wkT, wvT);
  k_prep_wT<<<(CDIM * CDIM + 255) / 256, 256, 0, stream>>>(Wo, woT, CDIM, CDIM);
  k_prep_wT<<<(CDIM * CFF + 255) / 256, 256, 0, stream>>>(W1, w1T, CDIM, CFF);
  k_prep_wT<<<(CFF * CDIM + 255) / 256, 256, 0, stream>>>(W2, w2T, CFF, CDIM);

  // 2) LN1
  k_layernorm<<<NTOK / 8, 256, 0, stream>>>(x, g1, be1, h1);

  // 3) QKV projections (bf16 out, [b,t,h,d] layout)
  dim3 gC(NTOK / 128, CDIM / 64);
  k_gemm<true, false, false><<<gC, 256, 0, stream>>>(h1, wqT, nullptr, nullptr,
                                                     qb, CDIM, CDIM);
  k_gemm<true, false, false><<<gC, 256, 0, stream>>>(h1, wkT, nullptr, nullptr,
                                                     kb, CDIM, CDIM);
  k_gemm<true, false, false><<<gC, 256, 0, stream>>>(h1, wvT, nullptr, nullptr,
                                                     vb, CDIM, CDIM);

  // 4) fused causal attention per (b,h)
  k_attention<<<BATCH * NHEAD, 256, ATT_SMEM, stream>>>(qb, kb, vb, ob);

  // 5) x2 = x + o @ Wo + bo   (f32 out)
  k_gemm<false, false, true><<<gC, 256, 0, stream>>>(ob, woT, bo, x, x2,
                                                     CDIM, CDIM);

  // 6) LN2, FF1 (exact GELU), FF2 (+bias +residual) -> d_out f32
  k_layernorm<<<NTOK / 8, 256, 0, stream>>>(x2, g2, be2, h2);
  dim3 gF(NTOK / 128, CFF / 64);
  k_gemm<true, true, false><<<gF, 256, 0, stream>>>(h2, w1T, b1, nullptr, ab,
                                                    CFF, CDIM);
  k_gemm<false, false, true><<<gC, 256, 0, stream>>>(ab, w2T, b2, x2,
                                                     (float*)d_out, CDIM, CFF);
}
